// SparseMoEBlock_24180665876614
// MI455X (gfx1250) — compile-verified, matrix operand order checked by reading, not run on hardware
//
#include <hip/hip_runtime.h>
#include <hip/hip_bf16.h>
#include <stdint.h>

// Problem constants (fixed by the reference)
#define TT 2048    // tokens = B*S
#define DD 1024    // hidden dim
#define EE 8       // experts
#define II 4096    // expert intermediate
#define SIDIM 2048 // shared intermediate

// Per-wave macro tile: 64 rows (4 x 16 M-subtiles) x 16 cols, K step 32.
#define MB 64
// LDS panel: 32 K-rows x 128 N-cols fp32, padded row stride (floats)
#define PROW 132

typedef __attribute__((ext_vector_type(16))) __bf16 v16bf;
typedef __attribute__((ext_vector_type(8)))  float  v8f;

union AFrag { v16bf v; uint4 q[2]; };

// ---- WMMA helpers ------------------------------------------------------

__device__ __forceinline__ v8f wmma_bf16(v16bf a, v16bf b, v8f c) {
  // D = A(16x32 bf16) * B(32x16 bf16) + C(16x16 f32)
  return __builtin_amdgcn_wmma_f32_16x16x32_bf16(
      /*neg_a=*/false, a, /*neg_b=*/false, b,
      /*c_mod=*/(short)0, c, /*reuse_a=*/false, /*reuse_b=*/false);
}

// A fragment: 16x32 bf16 tile; lanes 0-15 / 16-31 hold the same M rows with
// different K halves per the ISA 16-bit A layout.
__device__ __forceinline__ v16bf load_a_row(const __bf16* rowp, int k0) {
  const int lane = threadIdx.x & 31;
  const int koff = (lane < 16) ? 0 : 8;
  AFrag f;
  f.q[0] = *(const uint4*)(rowp + k0 + koff);        // K = k0+koff .. +7
  f.q[1] = *(const uint4*)(rowp + k0 + 16 + koff);   // K = k0+16+koff .. +7
  return f.v;
}

// ---- Async panel staging (Global -> LDS, no data registers) ------------
// Panel = weights[k0 .. k0+31][nbase .. nbase+127], fp32 row-major, ld = ldb.
// 4 x global_load_async_to_lds_b128 per thread, tracked by ASYNCcnt.

__device__ __forceinline__ void stage_async(const float* __restrict__ w, int ldb,
                                            int k0, int nbase, float* sB) {
  const int tid = threadIdx.x;
#pragma unroll
  for (int t = 0; t < 4; ++t) {
    int task = tid + t * 256;     // 0..1023
    int k    = task >> 5;         // 0..31
    int c4   = task & 31;         // float4 column
    const float* g = w + (size_t)(k0 + k) * ldb + nbase + c4 * 4;
    uint32_t ldsb = (uint32_t)(uintptr_t)(sB + k * PROW + c4 * 4);
    asm volatile("global_load_async_to_lds_b128 %0, %1, off"
                 :: "v"(ldsb), "v"((uint64_t)(uintptr_t)g)
                 : "memory");
  }
}

__device__ __forceinline__ void wait_async_all() {
  asm volatile("s_wait_asynccnt 0x0" ::: "memory");
}

// B fragment from staged LDS panel: 16 strided ds_load_b32 + bf16 convert.
// col = this wave's 16-column slice inside the 128-wide panel.
__device__ __forceinline__ v16bf load_b_lds(const float* sB, int col) {
  const int lane = threadIdx.x & 31;
  const int koff = (lane < 16) ? 0 : 8;
  const float* p = sB + koff * PROW + col + (lane & 15);
  v16bf b;
#pragma unroll
  for (int j = 0; j < 8; ++j) {
    b[j]     = (__bf16)p[j * PROW];
    b[j + 8] = (__bf16)p[(j + 16) * PROW];
  }
  return b;
}

__device__ __forceinline__ float silu_f(float g) {
  // g * sigmoid(g) with HW rcp (avoid IEEE div expansion)
  return g * __builtin_amdgcn_rcpf(1.f + __expf(-g));
}

// ---- 1) fp32 -> bf16 activation convert --------------------------------

__global__ void cvt_kernel(const float* __restrict__ x, __bf16* __restrict__ xb) {
  int i = blockIdx.x * blockDim.x + threadIdx.x;  // grid exactly covers TT*DD
  xb[i] = (__bf16)x[i];
}

// ---- 2) router: logits, softmax, top-2 ---------------------------------

__global__ void router_kernel(const float* __restrict__ x, const float* __restrict__ gw,
                              int* __restrict__ topk_idx, float* __restrict__ topk_w) {
  const int wave = threadIdx.x >> 5;
  const int lane = threadIdx.x & 31;
  const int t = blockIdx.x * 8 + wave;
  const float* xr = x + (size_t)t * DD;
  float acc[EE];
#pragma unroll
  for (int e = 0; e < EE; ++e) acc[e] = 0.f;
  for (int d = lane; d < DD; d += 32) {
    float xv = xr[d];
#pragma unroll
    for (int e = 0; e < EE; ++e) acc[e] += xv * gw[e * DD + d];
  }
#pragma unroll
  for (int e = 0; e < EE; ++e) {
#pragma unroll
    for (int off = 16; off > 0; off >>= 1)
      acc[e] += __shfl_xor(acc[e], off, 32);
  }
  if (lane == 0) {
    float mx = acc[0];
#pragma unroll
    for (int e = 1; e < EE; ++e) mx = fmaxf(mx, acc[e]);
    float p[EE]; float den = 0.f;
#pragma unroll
    for (int e = 0; e < EE; ++e) { p[e] = __expf(acc[e] - mx); den += p[e]; }
    const float inv = __builtin_amdgcn_rcpf(den);
    int i0 = 0;
#pragma unroll
    for (int e = 1; e < EE; ++e) if (p[e] > p[i0]) i0 = e;
    int i1 = (i0 == 0) ? 1 : 0;
#pragma unroll
    for (int e = 0; e < EE; ++e) if (e != i0 && p[e] > p[i1]) i1 = e;
    topk_idx[t * 2 + 0] = i0; topk_w[t * 2 + 0] = p[i0] * inv;
    topk_idx[t * 2 + 1] = i1; topk_w[t * 2 + 1] = p[i1] * inv;
  }
}

// ---- 3) deterministic per-expert buckets + prefix offsets --------------
// meta layout: [0..7]=count, [8..15]=padded count (mult of MB), [16..24]=row offsets

__global__ void bucket_kernel(const int* __restrict__ topk_idx, const float* __restrict__ topk_w,
                              int* __restrict__ btok, float* __restrict__ bw,
                              int* __restrict__ meta) {
  const int e = threadIdx.x;
  if (e < EE) {
    int cnt = 0;
    for (int t = 0; t < TT; ++t) {
#pragma unroll
      for (int k = 0; k < 2; ++k) {
        if (topk_idx[t * 2 + k] == e) {
          btok[e * TT + cnt] = t;
          bw[e * TT + cnt]   = topk_w[t * 2 + k];
          ++cnt;
        }
      }
    }
    meta[e] = cnt;
    int pad = (cnt + (MB - 1)) & ~(MB - 1);
    for (int j = cnt; j < pad; ++j) { btok[e * TT + j] = 0; bw[e * TT + j] = 0.f; }
    meta[8 + e] = pad;
  }
  __syncthreads();
  if (threadIdx.x == 0) {
    int off = 0;
    for (int q = 0; q < EE; ++q) { meta[16 + q] = off; off += meta[8 + q]; }
    meta[16 + EE] = off;
  }
}

// ---- 4) shared expert: fused gate+up+SiLU (64x16 per wave) -------------

__global__ __launch_bounds__(256)
void shared_gateup_kernel(const __bf16* __restrict__ xb,
                          const float* __restrict__ wsg,
                          const float* __restrict__ wsu,
                          __bf16* __restrict__ hs) {
  __shared__ __align__(16) float sBg[32 * PROW];
  __shared__ __align__(16) float sBu[32 * PROW];
  const int wave = threadIdx.x >> 5;
  const int lane = threadIdx.x & 31;
  const int nbase = blockIdx.x * 128;
  const int col = wave * 16;
  const int m0 = blockIdx.y * MB;
  const __bf16* rowp[4];
#pragma unroll
  for (int i = 0; i < 4; ++i)
    rowp[i] = xb + (size_t)(m0 + i * 16 + (lane & 15)) * DD;
  v8f cg[4], cu[4];
  const v8f vz = {};
#pragma unroll
  for (int i = 0; i < 4; ++i) { cg[i] = vz; cu[i] = vz; }
  stage_async(wsg, SIDIM, 0, nbase, sBg);
  stage_async(wsu, SIDIM, 0, nbase, sBu);
  wait_async_all();
  __syncthreads();
  for (int k0 = 0; k0 < DD; k0 += 32) {
    v16bf bg = load_b_lds(sBg, col);        // frags to regs first
    v16bf bu = load_b_lds(sBu, col);
    __syncthreads();                         // all waves done reading panel
    if (k0 + 32 < DD) {                      // async next panel overlaps WMMAs
      stage_async(wsg, SIDIM, k0 + 32, nbase, sBg);
      stage_async(wsu, SIDIM, k0 + 32, nbase, sBu);
    }
#pragma unroll
    for (int i = 0; i < 4; ++i) {
      v16bf a = load_a_row(rowp[i], k0);
      cg[i] = wmma_bf16(a, bg, cg[i]);
      cu[i] = wmma_bf16(a, bu, cu[i]);
    }
    wait_async_all();
    __syncthreads();                         // new panel visible
  }
  const int nn = nbase + col + (lane & 15);
#pragma unroll
  for (int i = 0; i < 4; ++i) {
#pragma unroll
    for (int r = 0; r < 8; ++r) {
      int m = i * 16 + ((lane < 16) ? r : (8 + r));
      float hv = silu_f(cg[i][r]) * cu[i][r];
      hs[(size_t)(m0 + m) * SIDIM + nn] = (__bf16)hv;
    }
  }
}

// ---- 5) shared expert down-proj: writes (initializes) d_out ------------

__global__ __launch_bounds__(256)
void shared_down_kernel(const __bf16* __restrict__ hs,
                        const float* __restrict__ wsd,
                        float* __restrict__ out) {
  __shared__ __align__(16) float sB[32 * PROW];
  const int wave = threadIdx.x >> 5;
  const int lane = threadIdx.x & 31;
  const int nbase = blockIdx.x * 128;
  const int col = wave * 16;
  const int m0 = blockIdx.y * MB;
  const __bf16* rowp[4];
#pragma unroll
  for (int i = 0; i < 4; ++i)
    rowp[i] = hs + (size_t)(m0 + i * 16 + (lane & 15)) * SIDIM;
  v8f c[4];
  const v8f vz = {};
#pragma unroll
  for (int i = 0; i < 4; ++i) c[i] = vz;
  stage_async(wsd, DD, 0, nbase, sB);
  wait_async_all();
  __syncthreads();
  for (int k0 = 0; k0 < SIDIM; k0 += 32) {
    v16bf b = load_b_lds(sB, col);
    __syncthreads();
    if (k0 + 32 < SIDIM) stage_async(wsd, DD, k0 + 32, nbase, sB);
#pragma unroll
    for (int i = 0; i < 4; ++i) {
      v16bf a = load_a_row(rowp[i], k0);
      c[i] = wmma_bf16(a, b, c[i]);
    }
    wait_async_all();
    __syncthreads();
  }
  const int nn = nbase + col + (lane & 15);
#pragma unroll
  for (int i = 0; i < 4; ++i) {
#pragma unroll
    for (int r = 0; r < 8; ++r) {
      int m = i * 16 + ((lane < 16) ? r : (8 + r));
      out[(size_t)(m0 + m) * DD + nn] = c[i][r];
    }
  }
}

// ---- 6) routed experts: gathered fused gate+up+SiLU --------------------

__global__ __launch_bounds__(256)
void moe_gateup_kernel(const __bf16* __restrict__ xb,
                       const float* __restrict__ wg,
                       const float* __restrict__ wu,
                       const int* __restrict__ btok,
                       const int* __restrict__ meta,
                       __bf16* __restrict__ h) {
  const int e  = blockIdx.z;
  const int mt = blockIdx.y;
  if (mt * MB >= meta[8 + e]) return;  // beyond this expert's padded rows
  __shared__ __align__(16) float sBg[32 * PROW];
  __shared__ __align__(16) float sBu[32 * PROW];
  const int wave = threadIdx.x >> 5;
  const int lane = threadIdx.x & 31;
  const int nbase = blockIdx.x * 128;
  const int col = wave * 16;
  const int rowbase = meta[16 + e] + mt * MB;
  const __bf16* rowp[4];
#pragma unroll
  for (int i = 0; i < 4; ++i) {
    int tok = btok[e * TT + mt * MB + i * 16 + (lane & 15)];
    rowp[i] = xb + (size_t)tok * DD;
  }
  const float* wge = wg + (size_t)e * DD * II;
  const float* wue = wu + (size_t)e * DD * II;
  v8f cg[4], cu[4];
  const v8f vz = {};
#pragma unroll
  for (int i = 0; i < 4; ++i) { cg[i] = vz; cu[i] = vz; }
  stage_async(wge, II, 0, nbase, sBg);
  stage_async(wue, II, 0, nbase, sBu);
  wait_async_all();
  __syncthreads();
  for (int k0 = 0; k0 < DD; k0 += 32) {
    v16bf bg = load_b_lds(sBg, col);
    v16bf bu = load_b_lds(sBu, col);
    __syncthreads();
    if (k0 + 32 < DD) {
      stage_async(wge, II, k0 + 32, nbase, sBg);
      stage_async(wue, II, k0 + 32, nbase, sBu);
    }
#pragma unroll
    for (int i = 0; i < 4; ++i) {
      v16bf a = load_a_row(rowp[i], k0);
      cg[i] = wmma_bf16(a, bg, cg[i]);
      cu[i] = wmma_bf16(a, bu, cu[i]);
    }
    wait_async_all();
    __syncthreads();
  }
  const int nn = nbase + col + (lane & 15);
#pragma unroll
  for (int i = 0; i < 4; ++i) {
#pragma unroll
    for (int r = 0; r < 8; ++r) {
      int m = i * 16 + ((lane < 16) ? r : (8 + r));
      float hv = silu_f(cg[i][r]) * cu[i][r];
      h[(size_t)(rowbase + m) * II + nn] = (__bf16)hv;
    }
  }
}

// ---- 7) routed down-proj + weighted scatter into d_out -----------------

__global__ __launch_bounds__(256)
void moe_down_kernel(const __bf16* __restrict__ h,
                     const float* __restrict__ wd,
                     const int* __restrict__ btok,
                     const float* __restrict__ bw,
                     const int* __restrict__ meta,
                     float* __restrict__ out) {
  const int e  = blockIdx.z;
  const int mt = blockIdx.y;
  if (mt * MB >= meta[8 + e]) return;
  __shared__ __align__(16) float sB[32 * PROW];
  const int wave = threadIdx.x >> 5;
  const int lane = threadIdx.x & 31;
  const int nbase = blockIdx.x * 128;
  const int col = wave * 16;
  const int rowbase = meta[16 + e] + mt * MB;
  const __bf16* rowp[4];
#pragma unroll
  for (int i = 0; i < 4; ++i)
    rowp[i] = h + (size_t)(rowbase + i * 16 + (lane & 15)) * II;
  const float* wde = wd + (size_t)e * II * DD;
  v8f c[4];
  const v8f vz = {};
#pragma unroll
  for (int i = 0; i < 4; ++i) c[i] = vz;
  stage_async(wde, DD, 0, nbase, sB);
  wait_async_all();
  __syncthreads();
  for (int k0 = 0; k0 < II; k0 += 32) {
    v16bf b = load_b_lds(sB, col);
    __syncthreads();
    if (k0 + 32 < II) stage_async(wde, DD, k0 + 32, nbase, sB);
#pragma unroll
    for (int i = 0; i < 4; ++i) {
      v16bf a = load_a_row(rowp[i], k0);
      c[i] = wmma_bf16(a, b, c[i]);
    }
    wait_async_all();
    __syncthreads();
  }
  const int nn = nbase + col + (lane & 15);
#pragma unroll
  for (int i = 0; i < 4; ++i) {
#pragma unroll
    for (int r = 0; r < 8; ++r) {
      int m = i * 16 + ((lane < 16) ? r : (8 + r));
      int   tok = btok[e * TT + mt * MB + m];
      float w   = bw[e * TT + mt * MB + m];
      atomicAdd(&out[(size_t)tok * DD + nn], c[i][r] * w);
    }
  }
}

// ---- launch ------------------------------------------------------------

extern "C" void kernel_launch(void* const* d_in, const int* in_sizes, int n_in,
                              void* d_out, int out_size, void* d_ws, size_t ws_size,
                              hipStream_t stream) {
  const float* x   = (const float*)d_in[0];
  const float* gw  = (const float*)d_in[1];
  const float* wg  = (const float*)d_in[2];
  const float* wu  = (const float*)d_in[3];
  const float* wd  = (const float*)d_in[4];
  const float* wsg = (const float*)d_in[5];
  const float* wsu = (const float*)d_in[6];
  const float* wsd = (const float*)d_in[7];
  float* out = (float*)d_out;

  char* ws = (char*)d_ws;
  int*    topk_idx = (int*)(ws + 0);                 // 16 KB
  float*  topk_w   = (float*)(ws + 16384);           // 16 KB
  int*    btok     = (int*)(ws + 32768);             // 64 KB
  float*  bw       = (float*)(ws + 98304);           // 64 KB
  int*    meta     = (int*)(ws + 163840);            // counts/padded/offsets
  __bf16* xb       = (__bf16*)(ws + (size_t)1  * (1 << 20)); // 4 MB
  __bf16* hs       = (__bf16*)(ws + (size_t)5  * (1 << 20)); // 8 MB
  __bf16* h        = (__bf16*)(ws + (size_t)13 * (1 << 20)); // <= ~38 MB

  cvt_kernel<<<TT * DD / 256, 256, 0, stream>>>(x, xb);
  router_kernel<<<TT / 8, 256, 0, stream>>>(x, gw, topk_idx, topk_w);
  bucket_kernel<<<1, 32, 0, stream>>>(topk_idx, topk_w, btok, bw, meta);
  shared_gateup_kernel<<<dim3(SIDIM / 128, TT / MB), 256, 0, stream>>>(xb, wsg, wsu, hs);
  shared_down_kernel<<<dim3(DD / 128, TT / MB), 256, 0, stream>>>(hs, wsd, out);
  moe_gateup_kernel<<<dim3(II / 128, TT / MB, EE), 256, 0, stream>>>(xb, wg, wu, btok, meta, h);
  moe_down_kernel<<<dim3(DD / 128, TT / MB, EE), 256, 0, stream>>>(h, wd, btok, bw, meta, out);
}
